// LinearAttention_83648783057407
// MI455X (gfx1250) — compile-verified, matrix operand order checked by reading, not run on hardware
//
#include <hip/hip_runtime.h>
#include <hip/hip_bf16.h>
#include <stdint.h>

// Problem constants (match reference exactly; lengths[] is statically full(L))
#define N_EV   65536   // G*L
#define IN_F   256
#define H_F    512
#define L_EV   128
#define CH     256                 // scan chunk length
#define NCHUNK (N_EV / CH)         // 256
#define LN_EPS 1e-5f

typedef __attribute__((ext_vector_type(16))) __bf16        v16bf;
typedef __attribute__((ext_vector_type(8)))  float         v8f;
typedef __attribute__((ext_vector_type(4)))  unsigned int  u32x4;
typedef __attribute__((ext_vector_type(4)))  unsigned int  uv4;
typedef __attribute__((ext_vector_type(8)))  int           iv8;
typedef __attribute__((ext_vector_type(4)))  int           iv4;
typedef int gv4i __attribute__((vector_size(16)));   // matches async builtin's pointee

#define AS1 __attribute__((address_space(1)))
#define AS3 __attribute__((address_space(3)))

// CDNA5 data-mover feature detection (device pass only; host pass and any
// toolchain lacking the builtins falls back to plain vector loads).
#if defined(__gfx1250__)
  #if __has_builtin(__builtin_amdgcn_tensor_load_to_lds) && __has_builtin(__builtin_amdgcn_s_wait_tensorcnt)
    #define USE_TDM 1
    #if __clang_major__ >= 23
      #define TDM_6ARG 1          // amdgpu-toolchain (clang-23+): 6-arg form
    #endif
  #endif
  #if __has_builtin(__builtin_amdgcn_global_load_async_to_lds_b128) && __has_builtin(__builtin_amdgcn_s_wait_asynccnt)
    #define USE_ASYNC 1
  #endif
#endif

union FragU { v16bf v; u32x4 u[2]; };

// float -> bf16 (round-to-nearest-even) without scalar __bf16 arithmetic
__device__ __forceinline__ unsigned short f2bf(float x) {
  unsigned u = __float_as_uint(x);
  unsigned r = u + 0x7FFFu + ((u >> 16) & 1u);
  return (unsigned short)(r >> 16);
}

// Load a 16x32 bf16 A-fragment row (ISA 7.12.2 layout):
// lanes 0-15 hold K {0..7, 16..23}, lanes 16-31 hold K {8..15, 24..31}.
// LDS rows are 40 elements (80 bytes): 16B-aligned and bank-conflict-free.
__device__ __forceinline__ v16bf load_frag(const unsigned short* rowBase, int hlf) {
  FragU f;
  const u32x4* p = (const u32x4*)rowBase;
  f.u[0] = p[hlf];       // K 0..7   (or 8..15)
  f.u[1] = p[2 + hlf];   // K 16..23 (or 24..31)
  return f.v;
}

// ---------------------------------------------------------------------------
// Generic C[N,512] = act(A[N,K] @ W[K,512]) GEMM, bf16 WMMA, f32 accumulate.
// Block tile 128x128, 8 waves (4x2), wave tile 32x64 (2x4 of 16x16), BK=32.
// A rows (optionally gathered through rowIdx) are staged via async
// global->LDS DMA; the weight tile is staged via the Tensor Data Mover.
// ---------------------------------------------------------------------------
__global__ __launch_bounds__(256)
void gemm_wmma(const float* __restrict__ A, const float* __restrict__ W,
               float* __restrict__ C, const int* __restrict__ rowIdx,
               int K, int act)
{
  __shared__ unsigned short lA[128 * 40];   // [m][k] bf16, stride 40
  __shared__ unsigned short lB[128 * 40];   // [n][k] bf16 (transposed), stride 40
#ifdef USE_TDM
  __shared__ float wstage[32 * 128];        // raw fp32 weight tile (TDM dest)
#endif
#ifdef USE_ASYNC
  __shared__ float astage[128 * 32];        // raw fp32 A tile (async-DMA dest)
#endif

  const int tid  = threadIdx.x;
  const int bn   = blockIdx.x, bm = blockIdx.y;
  const int lane = tid & 31,  wid = tid >> 5;
  const int hlf  = lane >> 4, idx = lane & 15;
  const int mw   = wid & 3,   nw  = wid >> 2;      // 4 (M) x 2 (N) waves

  // Hoist the (possibly gathered) A-row pointers: they are k-loop invariant.
  const float* aSrc[4];
  #pragma unroll
  for (int it = 0; it < 4; ++it) {
    int lin = tid + (it << 8);               // 1024 float4 slots, 128x8
    int r = lin >> 3, c4 = (lin & 7) << 2;
    int row = (bm << 7) + r;
    if (rowIdx) row = rowIdx[row];
    aSrc[it] = A + (size_t)row * K + c4;
  }

  v8f acc[2][4] = {};

  const int nk = K >> 5;
  for (int kt = 0; kt < nk; ++kt) {
    const int k0 = kt << 5;

    // ---------------- issue phase ----------------
#ifdef USE_TDM
    // Tensor Data Mover: DMA the 32x128 fp32 weight tile into LDS.
    if (wid == 0) {
      const unsigned lds_off = (unsigned)(size_t)(void*)wstage;   // addr[31:0] = LDS offset
      const unsigned long long ga =
          (unsigned long long)(size_t)(const void*)(W + (size_t)k0 * H_F + (bn << 7));
      uv4 g0;
      g0[0] = 1u;                                   // count=1, user descriptor
      g0[1] = lds_off;                              // lds_addr
      g0[2] = (unsigned)(ga & 0xFFFFFFFFull);       // global_addr[31:0]
      g0[3] = (unsigned)((ga >> 32) & 0x01FFFFFFull) | (2u << 30);  // addr[56:32] | type=2
      iv8 g1;
      g1[0] = (int)(2u << 16);                      // data_size = 4B
      g1[1] = (int)((unsigned)H_F << 16);           // tensor_dim0 = 512 (lo16 @ bits 63:48)
      g1[2] = (int)((unsigned)K << 16);             // tensor_dim1 = K   (lo16 @ bits 95:80)
      g1[3] = (int)(128u << 16);                    // tile_dim0 = 128   (bits 127:112)
      g1[4] = 32;                                   // tile_dim1 = 32    (bits 143:128)
      g1[5] = H_F;                                  // tensor_dim0_stride = 512
      g1[6] = 0; g1[7] = 0;
      iv4 z4 = {0, 0, 0, 0};                        // groups 2/3 unused (2-D tile)
#ifdef TDM_6ARG
      iv8 z8 = {0, 0, 0, 0, 0, 0, 0, 0};
      __builtin_amdgcn_tensor_load_to_lds(g0, g1, z4, z4, z8, 0);
#else
      __builtin_amdgcn_tensor_load_to_lds(g0, g1, z4, z4, 0);
#endif
    }
#endif

#ifdef USE_ASYNC
    // Async global->LDS DMA of the (gathered) fp32 A tile, 16B per lane slot.
    #pragma unroll
    for (int it = 0; it < 4; ++it) {
      int lin = tid + (it << 8);
      __builtin_amdgcn_global_load_async_to_lds_b128(
          (AS1 gv4i*)(uintptr_t)(aSrc[it] + k0),
          (AS3 gv4i*)(unsigned)(uintptr_t)(void*)&astage[(size_t)lin << 2],
          0, 0);
    }
#else
    float4 fa[4];
    #pragma unroll
    for (int it = 0; it < 4; ++it) fa[it] = *(const float4*)(aSrc[it] + k0);
#endif

#ifndef USE_TDM
    float4 fb[4];
    #pragma unroll
    for (int it = 0; it < 4; ++it) {
      int lin = tid + (it << 8);
      int kr = lin >> 5, c4 = (lin & 31) << 2;
      fb[it] = *(const float4*)(W + (size_t)(k0 + kr) * H_F + (bn << 7) + c4);
    }
#endif

    if (kt + 1 < nk) {  // hint next A k-slab into cache (global_prefetch_b8)
      __builtin_prefetch(aSrc[0] + k0 + 32, 0, 1);
      __builtin_prefetch(aSrc[2] + k0 + 32, 0, 1);
    }

    // ---------------- wait + convert phase ----------------
#ifdef USE_ASYNC
    __builtin_amdgcn_s_wait_asynccnt(0);    // own slots ready (self-read only)
#endif
#ifdef USE_TDM
    if (wid == 0) __builtin_amdgcn_s_wait_tensorcnt(0);
    __syncthreads();                        // publish wstage to all waves
#endif

    // A tile -> bf16 [m][k] stride 40
    #pragma unroll
    for (int it = 0; it < 4; ++it) {
      int lin = tid + (it << 8);
      int r = lin >> 3, c4 = (lin & 7) << 2;
#ifdef USE_ASYNC
      const float4 f = *(const float4*)&astage[(size_t)lin << 2];
#else
      const float4 f = fa[it];
#endif
      unsigned short* d = &lA[r * 40 + c4];
      d[0] = f2bf(f.x); d[1] = f2bf(f.y); d[2] = f2bf(f.z); d[3] = f2bf(f.w);
    }
    // W tile -> transposed bf16 [n][k] stride 40
    #pragma unroll
    for (int it = 0; it < 4; ++it) {
      int lin = tid + (it << 8);
      int kr = lin >> 5, c4 = (lin & 31) << 2;
#ifdef USE_TDM
      const float4 f = *(const float4*)&wstage[kr * 128 + c4];
#else
      const float4 f = fb[it];
#endif
      lB[(c4 + 0) * 40 + kr] = f2bf(f.x);
      lB[(c4 + 1) * 40 + kr] = f2bf(f.y);
      lB[(c4 + 2) * 40 + kr] = f2bf(f.z);
      lB[(c4 + 3) * 40 + kr] = f2bf(f.w);
    }
    __syncthreads();

    // ---------------- WMMA phase ----------------
    v16bf af[2], bfr[4];
    #pragma unroll
    for (int ms = 0; ms < 2; ++ms)
      af[ms] = load_frag(&lA[((mw << 5) + (ms << 4) + idx) * 40], hlf);
    #pragma unroll
    for (int ns = 0; ns < 4; ++ns)
      bfr[ns] = load_frag(&lB[((nw << 6) + (ns << 4) + idx) * 40], hlf);

    #pragma unroll
    for (int ms = 0; ms < 2; ++ms)
      #pragma unroll
      for (int ns = 0; ns < 4; ++ns)
        acc[ms][ns] = __builtin_amdgcn_wmma_f32_16x16x32_bf16(
            false, af[ms], false, bfr[ns], (short)0, acc[ms][ns], false, false);
    __syncthreads();
  }

  // Epilogue: C/D layout = VGPR r -> row r + 8*hlf, lane idx -> col.
  #pragma unroll
  for (int ms = 0; ms < 2; ++ms)
    #pragma unroll
    for (int ns = 0; ns < 4; ++ns)
      #pragma unroll
      for (int r = 0; r < 8; ++r) {
        int row = (bm << 7) + (mw << 5) + (ms << 4) + r + (hlf << 3);
        int col = (bn << 7) + (nw << 6) + (ns << 4) + idx;
        float vv = acc[ms][ns][r];
        if (act) vv = 1.0f / (1.0f + __expf(-vv));   // sigmoid for the gate
        C[(size_t)row * H_F + col] = vv;
      }
}

// ---------------------------------------------------------------------------
// Sort: 53-bit composite key (bid<<49 | orderable(time)<<17 | idx).
// Bitonic over 65536 keys == reference's stable (time, then bid) sort.
// ---------------------------------------------------------------------------
__global__ void k_build_keys(const float* __restrict__ time_,
                             const int* __restrict__ batch_id,
                             unsigned long long* __restrict__ keys) {
  int i = blockIdx.x * blockDim.x + threadIdx.x;
  if (i >= N_EV) return;
  unsigned bid = (unsigned)batch_id[i >> 7] & 0xFu;     // L_EV == 128
  unsigned tb = __float_as_uint(time_[i]);
  tb = (tb & 0x80000000u) ? ~tb : (tb | 0x80000000u);   // order-preserving
  keys[i] = ((unsigned long long)bid << 49) |
            ((unsigned long long)tb << 17) | (unsigned)i;
}

__global__ void k_bitonic(unsigned long long* __restrict__ keys, int j, int k) {
  int i = blockIdx.x * blockDim.x + threadIdx.x;
  int ixj = i ^ j;
  if (ixj <= i || ixj >= N_EV) return;
  unsigned long long a = keys[i], b = keys[ixj];
  bool up = ((i & k) == 0);
  if ((a > b) == up) { keys[i] = b; keys[ixj] = a; }
}

__global__ void k_sidx(const unsigned long long* __restrict__ keys,
                       int* __restrict__ sidx) {
  int i = blockIdx.x * blockDim.x + threadIdx.x;
  if (i < N_EV) sidx[i] = (int)(keys[i] & 0x1FFFFu);
}

// ---------------------------------------------------------------------------
// Segmented gated scan as plain scan with g_eff = start ? 0 : g.
// 3-phase chunked scan: per-chunk (prodG, partial), serial chunk combine,
// then re-apply with carry; phase 3 fuses qh = q*h in place.
// ---------------------------------------------------------------------------
__device__ __forceinline__ bool seg_start(const unsigned long long* keys, int i) {
  return (i == 0) || ((keys[i] >> 49) != (keys[i - 1] >> 49));
}

__global__ void k_scan_pass1(const unsigned long long* __restrict__ keys,
                             const float* __restrict__ g,
                             const float* __restrict__ v,
                             float* __restrict__ P, float* __restrict__ S) {
  int c = threadIdx.x;              // 512 channels
  int j = blockIdx.x;               // chunk
  float p = 1.f, s = 0.f;
  int base = j * CH;
  for (int t = 0; t < CH; ++t) {
    int i = base + t;
    float ge = seg_start(keys, i) ? 0.f : g[(size_t)i * H_F + c];
    p *= ge;
    s = ge * s + v[(size_t)i * H_F + c];
  }
  P[(size_t)j * H_F + c] = p;
  S[(size_t)j * H_F + c] = s;
}

__global__ void k_scan_combine(const float* __restrict__ P,
                               const float* __restrict__ S,
                               float* __restrict__ Cin) {
  int c = blockIdx.x * blockDim.x + threadIdx.x;
  if (c >= H_F) return;
  float carry = 0.f;
  for (int j = 0; j < NCHUNK; ++j) {
    Cin[(size_t)j * H_F + c] = carry;
    carry = P[(size_t)j * H_F + c] * carry + S[(size_t)j * H_F + c];
  }
}

__global__ void k_scan_pass3(const unsigned long long* __restrict__ keys,
                             const float* __restrict__ g,
                             const float* __restrict__ v,
                             float* __restrict__ q,        // in: q, out: q*h
                             const float* __restrict__ Cin) {
  int c = threadIdx.x;
  int j = blockIdx.x;
  float h = Cin[(size_t)j * H_F + c];
  int base = j * CH;
  for (int t = 0; t < CH; ++t) {
    int i = base + t;
    float ge = seg_start(keys, i) ? 0.f : g[(size_t)i * H_F + c];
    h = ge * h + v[(size_t)i * H_F + c];
    q[(size_t)i * H_F + c] *= h;
  }
}

// ---------------------------------------------------------------------------
// Fused un-permute + residual + LayerNorm. One wave32 per row of 512.
// ---------------------------------------------------------------------------
__global__ __launch_bounds__(256)
void k_ln_out(const float* __restrict__ oS, const float* __restrict__ res,
              const unsigned long long* __restrict__ keys,
              const float* __restrict__ gamma, const float* __restrict__ beta,
              float* __restrict__ out) {
  int lane = threadIdx.x & 31;
  int wid  = threadIdx.x >> 5;
  int i    = blockIdx.x * 8 + wid;                 // sorted row
  int orig = (int)(keys[i] & 0x1FFFFu);            // original row
  float vals[16];
  float s1 = 0.f, s2 = 0.f;
  #pragma unroll
  for (int e = 0; e < 16; ++e) {
    int c = (e << 5) + lane;
    float t = oS[(size_t)i * H_F + c] + res[(size_t)orig * H_F + c];
    vals[e] = t; s1 += t; s2 += t * t;
  }
  #pragma unroll
  for (int off = 16; off >= 1; off >>= 1) {
    s1 += __shfl_xor(s1, off, 32);
    s2 += __shfl_xor(s2, off, 32);
  }
  float mu  = s1 * (1.0f / H_F);
  float var = s2 * (1.0f / H_F) - mu * mu;
  float inv = rsqrtf(var + LN_EPS);
  #pragma unroll
  for (int e = 0; e < 16; ++e) {
    int c = (e << 5) + lane;
    out[(size_t)orig * H_F + c] = (vals[e] - mu) * inv * gamma[c] + beta[c];
  }
}

// ---------------------------------------------------------------------------
extern "C" void kernel_launch(void* const* d_in, const int* in_sizes, int n_in,
                              void* d_out, int out_size, void* d_ws, size_t ws_size,
                              hipStream_t stream) {
  (void)in_sizes; (void)n_in; (void)out_size; (void)ws_size;
  const float* events   = (const float*)d_in[0];
  const float* time_    = (const float*)d_in[1];
  const int*   batch_id = (const int*)d_in[2];
  /* d_in[3] = lengths: statically full(L) in the reference */
  const float* Win   = (const float*)d_in[4];
  const float* Wres  = (const float*)d_in[5];
  const float* Wq    = (const float*)d_in[6];
  const float* Wv    = (const float*)d_in[7];
  const float* Wg    = (const float*)d_in[8];
  const float* Wo    = (const float*)d_in[9];
  const float* gamma = (const float*)d_in[10];
  const float* beta  = (const float*)d_in[11];
  float* out = (float*)d_out;

  char* wp = (char*)d_ws;
  auto walloc = [&](size_t bytes) -> char* {
    char* p = wp; wp += (bytes + 255) & ~(size_t)255; return p;
  };
  const size_t NH = (size_t)N_EV * H_F * sizeof(float);
  unsigned long long* keys = (unsigned long long*)walloc(N_EV * 8);
  int*   sidx = (int*)  walloc(N_EV * 4);
  float* x    = (float*)walloc(NH);     // events@Win; recycled as oS later
  float* res  = (float*)walloc(NH);
  float* q    = (float*)walloc(NH);     // becomes q*h after pass3
  float* v    = (float*)walloc(NH);
  float* g    = (float*)walloc(NH);
  float* P    = (float*)walloc((size_t)NCHUNK * H_F * 4);
  float* S    = (float*)walloc((size_t)NCHUNK * H_F * 4);
  float* Cin  = (float*)walloc((size_t)NCHUNK * H_F * 4);
  float* oS   = x;                      // x dead after the q/v/g GEMMs

  // 1) sort keys
  k_build_keys<<<N_EV / 256, 256, 0, stream>>>(time_, batch_id, keys);
  for (int k = 2; k <= N_EV; k <<= 1)
    for (int j = k >> 1; j > 0; j >>= 1)
      k_bitonic<<<N_EV / 256, 256, 0, stream>>>(keys, j, k);
  k_sidx<<<N_EV / 256, 256, 0, stream>>>(keys, sidx);

  // 2) GEMMs (bf16 WMMA, f32 accumulate); gather fuses the permutation
  dim3 gg(H_F / 128, N_EV / 128);
  gemm_wmma<<<gg, 256, 0, stream>>>(events, Win,  x,   nullptr, IN_F, 0);
  gemm_wmma<<<gg, 256, 0, stream>>>(events, Wres, res, nullptr, IN_F, 0);
  gemm_wmma<<<gg, 256, 0, stream>>>(x, Wq, q, sidx, H_F, 0);
  gemm_wmma<<<gg, 256, 0, stream>>>(x, Wv, v, sidx, H_F, 0);
  gemm_wmma<<<gg, 256, 0, stream>>>(x, Wg, g, sidx, H_F, 1);   // sigmoid

  // 3) segmented gated scan (chunked), fusing qh = q*h
  k_scan_pass1  <<<NCHUNK, H_F, 0, stream>>>(keys, g, v, P, S);
  k_scan_combine<<<H_F / 256, 256, 0, stream>>>(P, S, Cin);
  k_scan_pass3  <<<NCHUNK, H_F, 0, stream>>>(keys, g, v, q, Cin);

  // 4) o = (q*h) @ Wo  (sorted order), then un-permute + residual + LN
  gemm_wmma<<<gg, 256, 0, stream>>>(q, Wo, oS, nullptr, H_F, 0);
  k_ln_out<<<N_EV / 8, 256, 0, stream>>>(oS, res, keys, gamma, beta, out);
}
